// transformer_71554155151471
// MI455X (gfx1250) — matrix operand order repaired and checked by reading, not compiled, not run on hardware
//
#include <hip/hip_runtime.h>
#include <cmath>

#define HID  300
#define HDIM 64
#define NH   8
#define DH   8
#define NL   6
#define BB   32
#define SS   512
#define CC   1000
#define MM   (BB*SS)      /* 16384 rows */
#define KP_HID 320        /* 300 padded to mult of 32 */
#define FF   1200
#define KP_FF 1216        /* 1200 padded to mult of 32 */

typedef __bf16 v16bf __attribute__((ext_vector_type(16)));
typedef float  v8f   __attribute__((ext_vector_type(8)));
typedef unsigned int u32x4 __attribute__((ext_vector_type(4)));

union FragBF { v16bf v; u32x4 q[2]; };

__device__ __forceinline__ unsigned short f2bf(float f) {
  union { float f; unsigned u; } v; v.f = f;
  unsigned r = v.u + 0x7FFFu + ((v.u >> 16) & 1u);   // round-to-nearest-even
  return (unsigned short)(r >> 16);
}

__device__ __forceinline__ v8f wmma_bf16(FragBF a, FragBF b, v8f c) {
  return __builtin_amdgcn_wmma_f32_16x16x32_bf16(
      false, a.v, false, b.v, (short)0, c, false, false);
}

// ---------------- embedding + positional encoding ----------------
// out = x1 + pe ; acc = 2*x1 + pe  (dense residual running sum init)
__global__ void embed_pe(const int* __restrict__ tok, const float* __restrict__ emb,
                         float* __restrict__ acc, float* __restrict__ outb) {
  size_t idx = (size_t)blockIdx.x * blockDim.x + threadIdx.x;
  if (idx >= (size_t)MM * HID) return;
  int f = (int)(idx % HID);
  size_t r = idx / HID;          // r = b*S + s
  int s = (int)(r & (SS - 1));
  float x1 = emb[(size_t)tok[r] * HID + f];
  float expo = (float)(f & ~1) / (float)HID;
  float ang = (float)s * __powf(10000.0f, -expo);
  float pe = (f & 1) ? __cosf(ang) : __sinf(ang);
  outb[idx] = x1 + pe;
  acc[idx]  = 2.0f * x1 + pe;
}

// ---------------- conversions ----------------
// f32 (M x K) row-major -> bf16 (M x Kp) row-major, zero pad
__global__ void cvt_a(const float* __restrict__ src, unsigned short* __restrict__ dst,
                      int K, int Kp, size_t total) {
  size_t idx = (size_t)blockIdx.x * blockDim.x + threadIdx.x;
  if (idx >= total) return;
  int k = (int)(idx % Kp);
  size_t m = idx / Kp;
  dst[idx] = (k < K) ? f2bf(src[m * (size_t)K + k]) : (unsigned short)0;
}

// weights f32 (K x N) row-major -> bf16 Bt (Npad x Kp), Bt[n][k] = W[k][n], zero pad
__global__ void cvt_bt(const float* __restrict__ w, unsigned short* __restrict__ dst,
                       int K, int N, int Kp, size_t total) {
  size_t idx = (size_t)blockIdx.x * blockDim.x + threadIdx.x;
  if (idx >= total) return;
  int k = (int)(idx % Kp);
  size_t n = idx / Kp;
  dst[idx] = (k < K && (int)n < N) ? f2bf(w[(size_t)k * N + n]) : (unsigned short)0;
}

__global__ void zero_pad(unsigned short* __restrict__ p, int ld, int cstart, size_t total) {
  size_t idx = (size_t)blockIdx.x * blockDim.x + threadIdx.x;
  if (idx >= total) return;
  int pad = ld - cstart;
  size_t r = idx / pad;
  int c = cstart + (int)(idx % pad);
  p[r * (size_t)ld + c] = 0;
}

// ---------------- register-blocked WMMA GEMM ----------------
// C(M x N) = A(M x Kp) @ Bt(Npad x Kp)^T ; wave computes a 64x64 block:
// 4 M-tiles x 4 N-tiles = 16 f32 accumulators (128 VGPRs). Per K-step: 8
// fragment loads feed 16 independent v_wmma ops (A reused 4x, B reused 4x).
template<int BIAS, int RELU, int OUTBF>
__global__ __launch_bounds__(32) void gemm_wmma(
    const unsigned short* __restrict__ A, const unsigned short* __restrict__ Bt,
    float* __restrict__ C, unsigned short* __restrict__ Cb,
    const float* __restrict__ bias, int Kp, int Mtot, int Nlog, int ldc)
{
  int lane = threadIdx.x;
  int mr = lane & 15, half = lane >> 4;
  size_t m0 = (size_t)blockIdx.y * 64;
  int n0 = blockIdx.x * 64;

  const unsigned short* arow[4];
  for (int i = 0; i < 4; ++i) {
    size_t r = m0 + i * 16 + mr;
    if (r >= (size_t)Mtot) r = (size_t)Mtot - 1;    // clamp (pad rows unused at store)
    arow[i] = A + r * (size_t)Kp;
  }
  const unsigned short* brow[4];
  for (int j = 0; j < 4; ++j)
    brow[j] = Bt + (size_t)(n0 + j * 16 + mr) * Kp + half * 16;

  v8f acc[4][4];
  for (int i = 0; i < 4; ++i)
    for (int j = 0; j < 4; ++j) acc[i][j] = (v8f){};

  for (int k = 0; k < Kp; k += 32) {
    FragBF fa[4], fb[4];
    for (int i = 0; i < 4; ++i) {
      fa[i].q[0] = *(const u32x4*)(arow[i] + k + half * 16);      // k[half*16, +8)
      fa[i].q[1] = *(const u32x4*)(arow[i] + k + half * 16 + 8);  // k[half*16+8, +8)
    }
    for (int j = 0; j < 4; ++j) {
      fb[j].q[0] = *(const u32x4*)(brow[j] + k);                  // k[half*16, +8)
      fb[j].q[1] = *(const u32x4*)(brow[j] + k + 8);              // k[half*16+8, +8)
    }
    for (int i = 0; i < 4; ++i)
      for (int j = 0; j < 4; ++j)
        acc[i][j] = wmma_bf16(fa[i], fb[j], acc[i][j]);
  }

  for (int j = 0; j < 4; ++j) {
    int col = n0 + j * 16 + mr;
    if (col >= Nlog) continue;
    float bv = BIAS ? bias[col] : 0.0f;
    for (int i = 0; i < 4; ++i) {
      for (int e = 0; e < 8; ++e) {
        size_t rowi = m0 + i * 16 + half * 8 + e;
        if (rowi >= (size_t)Mtot) break;
        float v = acc[i][j][e] + bv;
        if (RELU) v = fmaxf(v, 0.0f);
        if (OUTBF) Cb[rowi * (size_t)ldc + col] = f2bf(v);
        else       C [rowi * (size_t)ldc + col] = v;
      }
    }
  }
}

// ---------------- split fused QKV into per-head WMMA-ready bf16 layouts ----------------
// q/k feature index for (d,h) is d*NH+h (reference reshape). Pads d to 32 with zeros.
// vT stored as [hb][d(16)][s] for B-operand of alpha@V.
__global__ void qkv_split(const float* __restrict__ qkv,
                          unsigned short* __restrict__ qh, unsigned short* __restrict__ kh,
                          unsigned short* __restrict__ vt) {
  int idx = blockIdx.x * blockDim.x + threadIdx.x;
  if (idx >= NH * BB * SS) return;
  int s = idx & (SS - 1);
  int b = (idx >> 9) & (BB - 1);
  int h = idx >> 14;
  int hb = h * BB + b;
  const float* row = qkv + ((size_t)b * SS + s) * 192;
  unsigned short* qd = qh + ((size_t)hb * SS + s) * 32;
  unsigned short* kd = kh + ((size_t)hb * SS + s) * 32;
  for (int d = 0; d < 32; ++d) {
    qd[d] = (d < DH) ? f2bf(row[d * NH + h])      : (unsigned short)0;
    kd[d] = (d < DH) ? f2bf(row[64 + d * NH + h]) : (unsigned short)0;
  }
  for (int d = 0; d < 16; ++d)
    vt[((size_t)hb * 16 + d) * SS + s] =
        (d < DH) ? f2bf(row[128 + d * NH + h]) : (unsigned short)0;
}

// ---------------- fused attention: scores(WMMA) -> masked softmax(LDS) -> alpha@V(WMMA) ----
__global__ __launch_bounds__(32) void attn_kernel(
    const unsigned short* __restrict__ qh, const unsigned short* __restrict__ kh,
    const unsigned short* __restrict__ vt, const int* __restrict__ tok,
    float* __restrict__ attn, float scale)
{
  __shared__ float          sc[16 * SS];
  __shared__ unsigned short al[16 * SS];
  int lane = threadIdx.x;
  int mr = lane & 15, half = lane >> 4;
  int hb = blockIdx.y;
  int b = hb & (BB - 1);
  int h = hb >> 5;
  int q0 = blockIdx.x * 16;
  const unsigned short* qbase = qh + (size_t)hb * SS * 32;
  const unsigned short* kbase = kh + (size_t)hb * SS * 32;

  FragBF fq;
  const unsigned short* qrow = qbase + (size_t)(q0 + mr) * 32;
  fq.q[0] = *(const u32x4*)(qrow + half * 16);
  fq.q[1] = *(const u32x4*)(qrow + half * 16 + 8);

  for (int kt = 0; kt < SS / 16; ++kt) {
    int k0 = kt * 16;
    FragBF fk;
    const unsigned short* krow = kbase + (size_t)(k0 + mr) * 32 + half * 16;
    fk.q[0] = *(const u32x4*)(krow);
    fk.q[1] = *(const u32x4*)(krow + 8);
    v8f e = {};
    e = wmma_bf16(fq, fk, e);              // K=32 (DH padded) single step, exact
    int col = k0 + mr;
    bool msk = (tok[(size_t)b * SS + col] == 0);
    for (int i = 0; i < 8; ++i)
      sc[(half * 8 + i) * SS + col] = msk ? -1e9f : e[i] * scale;
  }
  __syncthreads();

  // softmax: lane handles row mr, column half [half*256, +256); combine halves via lane^16
  int c0 = half * 256;
  float mx = -3.0e38f;
  for (int c = 0; c < 256; ++c) mx = fmaxf(mx, sc[mr * SS + c0 + c]);
  mx = fmaxf(mx, __shfl_xor(mx, 16));
  float sum = 0.0f;
  for (int c = 0; c < 256; ++c) {
    float e = __expf(sc[mr * SS + c0 + c] - mx);
    sc[mr * SS + c0 + c] = e;
    sum += e;
  }
  sum += __shfl_xor(sum, 16);
  float inv = 1.0f / sum;
  for (int c = 0; c < 256; ++c)
    al[mr * SS + c0 + c] = f2bf(sc[mr * SS + c0 + c] * inv);
  __syncthreads();

  // alpha (16 x 512) @ V (512 x 16-padded): 16 WMMA k-steps
  v8f o = {};
  for (int ks = 0; ks < SS; ks += 32) {
    FragBF fa, fv;
    const unsigned short* arow = al + (size_t)mr * SS + ks;
    fa.q[0] = *(const u32x4*)(arow + half * 16);
    fa.q[1] = *(const u32x4*)(arow + half * 16 + 8);
    const unsigned short* vrow = vt + ((size_t)hb * 16 + mr) * SS + ks + half * 16;
    fv.q[0] = *(const u32x4*)(vrow);
    fv.q[1] = *(const u32x4*)(vrow + 8);
    o = wmma_bf16(fa, fv, o);
  }
  if (mr < DH) {
    for (int i = 0; i < 8; ++i) {
      int s = q0 + half * 8 + i;
      attn[((size_t)b * SS + s) * HDIM + h * DH + mr] = o[i];   // head-major concat
    }
  }
}

// ---------------- residual + LayerNorm (ddof=1), optional acc += out ----------------
template<int UPD>
__global__ __launch_bounds__(128) void ln_res(
    const float* __restrict__ xin, const float* __restrict__ res, float* __restrict__ outp,
    const float* __restrict__ gam, const float* __restrict__ bet, int l,
    float* __restrict__ accUpd)
{
  int lane = threadIdx.x & 31;
  int w = threadIdx.x >> 5;
  size_t row = (size_t)blockIdx.x * 4 + w;
  const float* a = xin + row * HID;
  const float* r = res + row * HID;
  float vals[10];
  float s = 0.0f, sq = 0.0f;
  for (int j = 0; j < 10; ++j) {
    int c = lane + j * 32;
    float v = (c < HID) ? a[c] + r[c] : 0.0f;
    vals[j] = v; s += v; sq += v * v;
  }
  for (int off = 16; off > 0; off >>= 1) {
    s  += __shfl_xor(s,  off);
    sq += __shfl_xor(sq, off);
  }
  float mu = s / (float)HID;
  float var = (sq - mu * s) / (float)(HID - 1);
  float rstd = rsqrtf(var + 1e-8f);
  float g = gam[l], be = bet[l];
  for (int j = 0; j < 10; ++j) {
    int c = lane + j * 32;
    if (c < HID) {
      float o = g * (vals[j] - mu) * rstd + be;
      outp[row * HID + c] = o;
      if (UPD) accUpd[row * HID + c] += o;
    }
  }
}

// ---------------- sum over sequence ----------------
__global__ void seq_sum(const float* __restrict__ outb, float* __restrict__ sums) {
  int idx = blockIdx.x * blockDim.x + threadIdx.x;
  if (idx >= BB * HID) return;
  int b = idx / HID, f = idx % HID;
  float s = 0.0f;
  for (int t = 0; t < SS; ++t) s += outb[((size_t)b * SS + t) * HID + f];
  sums[idx] = s;
}

// ---------------- final LayerNorm(ddof=1) + softmax over C=1000 ----------------
__global__ __launch_bounds__(256) void final_kernel(
    const float* __restrict__ logits, const float* __restrict__ gf,
    const float* __restrict__ bf, float* __restrict__ outp)
{
  __shared__ float red[256];
  int b = blockIdx.x, t = threadIdx.x;
  const float* row = logits + (size_t)b * CC;
  float* orow = outp + (size_t)b * CC;
  float g = gf[0], be = bf[0];

  auto reduce_sum = [&](float v) -> float {
    red[t] = v; __syncthreads();
    for (int o = 128; o > 0; o >>= 1) { if (t < o) red[t] += red[t + o]; __syncthreads(); }
    float r = red[0]; __syncthreads();
    return r;
  };
  auto reduce_max = [&](float v) -> float {
    red[t] = v; __syncthreads();
    for (int o = 128; o > 0; o >>= 1) { if (t < o) red[t] = fmaxf(red[t], red[t + o]); __syncthreads(); }
    float r = red[0]; __syncthreads();
    return r;
  };

  float s = 0.0f;
  for (int c = t; c < CC; c += 256) s += row[c];
  float mu = reduce_sum(s) / (float)CC;
  float sq = 0.0f;
  for (int c = t; c < CC; c += 256) { float d = row[c] - mu; sq += d * d; }
  float var = reduce_sum(sq) / (float)(CC - 1);
  float rstd = rsqrtf(var + 1e-8f);
  float mx = -3.0e38f;
  for (int c = t; c < CC; c += 256) {
    float y = g * (row[c] - mu) * rstd + be;
    orow[c] = y; mx = fmaxf(mx, y);
  }
  mx = reduce_max(mx);
  float es = 0.0f;
  for (int c = t; c < CC; c += 256) { float e = __expf(orow[c] - mx); orow[c] = e; es += e; }
  float tot = reduce_sum(es);
  float inv = 1.0f / tot;
  for (int c = t; c < CC; c += 256) orow[c] *= inv;
}

// =======================================================================
extern "C" void kernel_launch(void* const* d_in, const int* in_sizes, int n_in,
                              void* d_out, int out_size, void* d_ws, size_t ws_size,
                              hipStream_t stream) {
  (void)in_sizes; (void)n_in; (void)out_size; (void)ws_size;
  const int*   tok = (const int*)  d_in[0];
  const float* emb = (const float*)d_in[1];
  const float* WQ  = (const float*)d_in[2];
  const float* WK  = (const float*)d_in[3];
  const float* WV  = (const float*)d_in[4];
  const float* WO  = (const float*)d_in[5];
  const float* W1  = (const float*)d_in[6];
  const float* b1  = (const float*)d_in[7];
  const float* W2  = (const float*)d_in[8];
  const float* b2  = (const float*)d_in[9];
  const float* g1  = (const float*)d_in[10];
  const float* be1 = (const float*)d_in[11];
  const float* g2  = (const float*)d_in[12];
  const float* be2 = (const float*)d_in[13];
  const float* Vd  = (const float*)d_in[14];
  const float* gf  = (const float*)d_in[15];
  const float* bfp = (const float*)d_in[16];
  float* outp = (float*)d_out;

  char* base = (char*)d_ws;
  size_t off = 0;
  auto alloc = [&](size_t bytes) -> void* {
    void* p = base + off;
    off += (bytes + 255) & ~(size_t)255;
    return p;
  };
  float* acc   = (float*)alloc((size_t)MM * HID * 4);
  float* outb  = (float*)alloc((size_t)MM * HID * 4);
  float* xa    = (float*)alloc((size_t)MM * HID * 4);
  float* tmp   = (float*)alloc((size_t)MM * HID * 4);     // mha-proj out, then FF2 out
  float* attn  = (float*)alloc((size_t)MM * HDIM * 4);
  float* qkv   = (float*)alloc((size_t)MM * 192 * 4);
  unsigned short* f1b   = (unsigned short*)alloc((size_t)MM * KP_FF * 2);
  unsigned short* abuf  = (unsigned short*)alloc((size_t)MM * KP_HID * 2);
  unsigned short* btq   = (unsigned short*)alloc((size_t)192 * KP_HID * 2);
  unsigned short* bto   = (unsigned short*)alloc((size_t)320 * 64 * 2);
  unsigned short* bt1   = (unsigned short*)alloc((size_t)1216 * KP_HID * 2);
  unsigned short* bt2   = (unsigned short*)alloc((size_t)320 * KP_FF * 2);
  unsigned short* btv   = (unsigned short*)alloc((size_t)1024 * KP_HID * 2);
  unsigned short* qh    = (unsigned short*)alloc((size_t)NH * BB * SS * 32 * 2);
  unsigned short* kh    = (unsigned short*)alloc((size_t)NH * BB * SS * 32 * 2);
  unsigned short* vt    = (unsigned short*)alloc((size_t)NH * BB * 16 * SS * 2);
  float* sums  = (float*)alloc((size_t)BB * HID * 4);
  unsigned short* sumsb = (unsigned short*)alloc((size_t)BB * KP_HID * 2);
  float* logits = (float*)alloc((size_t)BB * CC * 4);

  const float scale = 1.0f / sqrtf((float)HID);

  embed_pe<<<dim3((unsigned)(((size_t)MM * HID + 255) / 256)), 256, 0, stream>>>(tok, emb, acc, outb);
  zero_pad<<<dim3((MM * 16 + 255) / 256), 256, 0, stream>>>(f1b, KP_FF, FF, (size_t)MM * 16);

  for (int l = 0; l < NL; ++l) {
    // ---- QKV: (16384 x 320) @ (320 x 192) ----
    cvt_a<<<dim3((unsigned)(((size_t)MM * KP_HID + 255) / 256)), 256, 0, stream>>>(
        acc, abuf, HID, KP_HID, (size_t)MM * KP_HID);
    const float* wsrc[3] = { WQ + (size_t)l * HID * HDIM,
                             WK + (size_t)l * HID * HDIM,
                             WV + (size_t)l * HID * HDIM };
    for (int p = 0; p < 3; ++p)
      cvt_bt<<<dim3((64 * KP_HID + 255) / 256), 256, 0, stream>>>(
          wsrc[p], btq + (size_t)p * 64 * KP_HID, HID, 64, KP_HID, (size_t)64 * KP_HID);
    gemm_wmma<0, 0, 0><<<dim3(3, MM / 64), 32, 0, stream>>>(
        abuf, btq, qkv, nullptr, nullptr, KP_HID, MM, 192, 192);

    // ---- attention ----
    qkv_split<<<dim3((NH * BB * SS + 255) / 256), 256, 0, stream>>>(qkv, qh, kh, vt);
    attn_kernel<<<dim3(SS / 16, NH * BB), 32, 0, stream>>>(qh, kh, vt, tok, attn, scale);

    // ---- output projection + LN1: (16384 x 64) @ (64 x 300) ----
    cvt_a<<<dim3((unsigned)(((size_t)MM * 64 + 255) / 256)), 256, 0, stream>>>(
        attn, abuf, 64, 64, (size_t)MM * 64);
    cvt_bt<<<dim3((320 * 64 + 255) / 256), 256, 0, stream>>>(
        WO + (size_t)l * HDIM * HID, bto, 64, HID, 64, (size_t)320 * 64);
    gemm_wmma<0, 0, 0><<<dim3(5, MM / 64), 32, 0, stream>>>(
        abuf, bto, tmp, nullptr, nullptr, 64, MM, HID, HID);
    ln_res<0><<<dim3(MM / 4), 128, 0, stream>>>(tmp, acc, xa, g1, be1, l, nullptr);

    // ---- FFN: (16384 x 320)@(320 x 1200) relu -> (16384 x 1216)@(1216 x 300) ----
    cvt_a<<<dim3((unsigned)(((size_t)MM * KP_HID + 255) / 256)), 256, 0, stream>>>(
        xa, abuf, HID, KP_HID, (size_t)MM * KP_HID);
    cvt_bt<<<dim3((unsigned)((1216 * (size_t)KP_HID + 255) / 256)), 256, 0, stream>>>(
        W1 + (size_t)l * HID * FF, bt1, HID, FF, KP_HID, (size_t)1216 * KP_HID);
    gemm_wmma<1, 1, 1><<<dim3(19, MM / 64), 32, 0, stream>>>(
        abuf, bt1, nullptr, f1b, b1 + (size_t)l * FF, KP_HID, MM, FF, KP_FF);
    cvt_bt<<<dim3((unsigned)((320 * (size_t)KP_FF + 255) / 256)), 256, 0, stream>>>(
        W2 + (size_t)l * FF * HID, bt2, FF, HID, KP_FF, (size_t)320 * KP_FF);
    gemm_wmma<1, 0, 0><<<dim3(5, MM / 64), 32, 0, stream>>>(
        f1b, bt2, tmp, nullptr, b2 + (size_t)l * HID, KP_FF, MM, HID, HID);
    ln_res<1><<<dim3(MM / 4), 128, 0, stream>>>(tmp, xa, outb, g2, be2, l, acc);
  }

  // ---- final: sum over S, logits GEMM (32 x 320)@(320 x 1000), LN + softmax ----
  seq_sum<<<dim3((BB * HID + 255) / 256), 256, 0, stream>>>(outb, sums);
  cvt_a<<<dim3((BB * KP_HID + 255) / 256), 256, 0, stream>>>(
      sums, sumsb, HID, KP_HID, (size_t)BB * KP_HID);
  cvt_bt<<<dim3((unsigned)((1024 * (size_t)KP_HID + 255) / 256)), 256, 0, stream>>>(
      Vd, btv, HID, CC, KP_HID, (size_t)1024 * KP_HID);
  gemm_wmma<0, 0, 0><<<dim3(16, 1), 32, 0, stream>>>(
      sumsb, btv, logits, nullptr, nullptr, KP_HID, BB, CC, CC);
  final_kernel<<<dim3(BB), 256, 0, stream>>>(logits, gf, bfp, outp);
}
